// MultiHeadSelfAttention_65833258713203
// MI455X (gfx1250) — compile-verified, matrix operand order checked by reading, not run on hardware
//
#include <hip/hip_runtime.h>

// ---------------------------------------------------------------------------
// MI455X (gfx1250) multi-head self-attention.
// bf16 WMMA everywhere (fp32 accumulate), async global->LDS staging for GEMM
// A-tiles, prefetch for next K-tile, flash attention with online softmax.
// ---------------------------------------------------------------------------

typedef __attribute__((ext_vector_type(16))) __bf16 v16bf;
typedef __attribute__((ext_vector_type(8)))  __bf16 v8bf;
typedef __attribute__((ext_vector_type(8)))  float  v8f;

#define S_LEN   4096
#define DMODEL  2048
#define NH      16
#define HD      128
#define KBLK    64      // keys per flash-attention block
#define QROWS   128     // query rows per workgroup (8 waves x 16)

// Load one 16x32 bf16 A/B fragment (ISA 7.12.2 layout) for this lane.
// lane half (lane>=16) selects klo = base + 8; elements are
// [klo..klo+7] and [klo+16..klo+23] of the lane's row -> two 16B ds loads.
__device__ __forceinline__ v16bf load_frag(const __bf16* rowptr, int klo) {
    v8bf lo = *(const v8bf*)(rowptr + klo);
    v8bf hi = *(const v8bf*)(rowptr + klo + 16);
    v16bf r;
#pragma unroll
    for (int i = 0; i < 8; ++i) { r[i] = lo[i]; r[i + 8] = hi[i]; }
    return r;
}

// ---------------------------------------------------------------------------
// fp32 -> bf16 elementwise conversion (one-time repack of x / weights)
// ---------------------------------------------------------------------------
__global__ __launch_bounds__(256)
void to_bf16(const float* __restrict__ in, __bf16* __restrict__ out, int n) {
    int i = blockIdx.x * 256 + threadIdx.x;
    if (i < n) out[i] = (__bf16)in[i];
}

// ---------------------------------------------------------------------------
// C[M,N] = A[M,K] * B[K,N] + bias;  A,B bf16, accumulate fp32, OutT output.
// Block tile 128x64, K-tile 64; 8 waves as 4x2, each wave 32x32.
// A tile staged with GLOBAL_LOAD_ASYNC_TO_LDS_B128 (ASYNCcnt path, no VGPR
// round-trip); B tile staged transposed (N-major) manually for fragment reads.
// ---------------------------------------------------------------------------
template <typename OutT>
__global__ __launch_bounds__(256)
void gemm_bias_bf16(const __bf16* __restrict__ A, const __bf16* __restrict__ B,
                    const float* __restrict__ bias, OutT* __restrict__ C,
                    int M, int N, int K) {
    __shared__ __align__(16) __bf16 As[128][72];   // 64 k + pad 8 (bank-safe)
    __shared__ __align__(16) __bf16 Bs[64][72];    // N-major: Bs[n][k]

    const int tid  = threadIdx.x;
    const int lane = tid & 31;
    const int wave = tid >> 5;
    const int half = lane >> 4;
    const int lr   = lane & 15;
    const int wm   = wave >> 1;      // 0..3 -> 32-row strips
    const int wn   = wave & 1;       // 0..1 -> 32-col strips
    const int bm   = blockIdx.y * 128;
    const int bn   = blockIdx.x * 64;

    v8f c[2][2];
#pragma unroll
    for (int mt = 0; mt < 2; ++mt)
#pragma unroll
        for (int nt = 0; nt < 2; ++nt)
#pragma unroll
            for (int v = 0; v < 8; ++v) c[mt][nt][v] = 0.0f;

    for (int k0 = 0; k0 < K; k0 += 64) {
        // ---- A tile 128x64 bf16: 1024 16B chunks, 4/thread, async -> LDS ----
#pragma unroll
        for (int i = 0; i < 4; ++i) {
            int ch  = tid + i * 256;
            int row = ch >> 3;
            int kk  = (ch & 7) * 8;
            unsigned lds = (unsigned)(size_t)&As[row][kk];
            unsigned long long ga =
                (unsigned long long)(size_t)(A + (size_t)(bm + row) * K + k0 + kk);
            asm volatile("global_load_async_to_lds_b128 %0, %1, off"
                         :: "v"(lds), "v"(ga) : "memory");
        }
        // ---- B tile 64x64 bf16 staged transposed: 512 chunks, 2/thread ----
#pragma unroll
        for (int i = 0; i < 2; ++i) {
            int ch   = tid + i * 256;
            int krow = ch >> 3;
            int nn   = (ch & 7) * 8;
            v8bf t = *(const v8bf*)(B + (size_t)(k0 + krow) * N + bn + nn);
#pragma unroll
            for (int j = 0; j < 8; ++j) Bs[nn + j][krow] = t[j];
        }
        // ---- prefetch next K-tile into cache (global_prefetch_b8) ----
        if (k0 + 64 < K) {
            __builtin_prefetch(A + (size_t)(bm + (tid >> 1)) * K + k0 + 64, 0, 0);
            __builtin_prefetch(B + (size_t)(k0 + 64 + (tid >> 2)) * N + bn, 0, 0);
        }
        asm volatile("s_wait_asynccnt 0x0" ::: "memory");
        __syncthreads();

#pragma unroll
        for (int kt = 0; kt < 2; ++kt) {
            const int klo = kt * 32 + half * 8;
            v16bf af[2], bf[2];
#pragma unroll
            for (int mt = 0; mt < 2; ++mt)
                af[mt] = load_frag(&As[wm * 32 + mt * 16 + lr][0], klo);
#pragma unroll
            for (int nt = 0; nt < 2; ++nt)
                bf[nt] = load_frag(&Bs[wn * 32 + nt * 16 + lr][0], klo);
#pragma unroll
            for (int mt = 0; mt < 2; ++mt)
#pragma unroll
                for (int nt = 0; nt < 2; ++nt)
                    c[mt][nt] = __builtin_amdgcn_wmma_f32_16x16x32_bf16(
                        false, af[mt], false, bf[nt], (short)0, c[mt][nt],
                        false, false);
        }
        __syncthreads();
    }

    // epilogue: C layout is M = v + 8*half, N = lane%16 (ISA 7.12.2)
#pragma unroll
    for (int mt = 0; mt < 2; ++mt)
#pragma unroll
        for (int nt = 0; nt < 2; ++nt)
#pragma unroll
            for (int v = 0; v < 8; ++v) {
                int row = bm + wm * 32 + mt * 16 + v + 8 * half;
                int col = bn + wn * 32 + nt * 16 + lr;
                C[(size_t)row * N + col] = (OutT)(c[mt][nt][v] + bias[col]);
            }
}

// ---------------------------------------------------------------------------
// RoPE + repack: qkv bf16 [S][3*D] -> head-major bf16 Q/K/V [H][S][HD].
// Score scale 1/sqrt(hd) folded into Q.
// ---------------------------------------------------------------------------
__global__ __launch_bounds__(256)
void rope_repack(const __bf16* __restrict__ qkv,
                 __bf16* __restrict__ Qb, __bf16* __restrict__ Kb,
                 __bf16* __restrict__ Vb) {
    int idx = blockIdx.x * 256 + threadIdx.x;        // over S*H*HD = 8M
    int d  = idx & (HD - 1);
    int sh = idx >> 7;
    int h  = sh & (NH - 1);
    int s  = sh >> 4;

    const __bf16* base = qkv + (size_t)s * (3 * DMODEL) + h * HD;
    float qv = (float)base[d];
    float kv = (float)base[DMODEL + d];
    float vv = (float)base[2 * DMODEL + d];
    float qo = qv, ko = kv;
    if (d < 64) {
        // inv_freq[i] = 10000^(-i/32);  angle = s * inv_freq[d%32]
        float invf = __expf(-0.28782323254f * (float)(d & 31));
        float ang  = (float)s * invf;
        float cs = __cosf(ang), sn = __sinf(ang);
        if (d < 32) {
            qo = qv * cs - (float)base[d + 32] * sn;
            ko = kv * cs - (float)base[DMODEL + d + 32] * sn;
        } else {
            qo = qv * cs + (float)base[d - 32] * sn;
            ko = kv * cs + (float)base[DMODEL + d - 32] * sn;
        }
    }
    size_t o = ((size_t)h * S_LEN + s) * HD + d;
    Qb[o] = (__bf16)(qo * 0.08838834764831845f);     // 1/sqrt(128)
    Kb[o] = (__bf16)ko;
    Vb[o] = (__bf16)vv;
}

// ---------------------------------------------------------------------------
// Flash attention: block = (head, 128 query rows); wave owns 16 rows.
// Per 64-key block: 16 WMMA (S=QK^T) + online softmax + 16 WMMA (PV).
// Output written directly as bf16 [S][D] for the final projection GEMM.
// ---------------------------------------------------------------------------
__global__ __launch_bounds__(256)
void attn_kernel(const __bf16* __restrict__ Qb, const __bf16* __restrict__ Kb,
                 const __bf16* __restrict__ Vb, __bf16* __restrict__ attnb) {
    __shared__ __align__(16) __bf16 Vs[HD][KBLK + 8];      // transposed V tile
    __shared__ __align__(16) __bf16 Ps[8][16][KBLK + 8];   // per-wave P tile

    const int tid  = threadIdx.x;
    const int lane = tid & 31;
    const int wave = tid >> 5;
    const int half = lane >> 4;
    const int lr   = lane & 15;
    const int h    = blockIdx.x / (S_LEN / QROWS);
    const int qb   = blockIdx.x % (S_LEN / QROWS);
    const int r0   = qb * QROWS + wave * 16;

    // Q fragments stay resident: 4 tiles over hd (K-steps of 32)
    v16bf qf[4];
    {
        const __bf16* qrow = Qb + ((size_t)h * S_LEN + r0 + lr) * HD;
#pragma unroll
        for (int kt = 0; kt < 4; ++kt)
            qf[kt] = load_frag(qrow, kt * 32 + half * 8);
    }

    float m_i[8], l_i[8];
    v8f o[8];
#pragma unroll
    for (int v = 0; v < 8; ++v) { m_i[v] = -3.0e38f; l_i[v] = 0.0f; }
#pragma unroll
    for (int t = 0; t < 8; ++t)
#pragma unroll
        for (int v = 0; v < 8; ++v) o[t][v] = 0.0f;

    for (int kb = 0; kb < S_LEN / KBLK; ++kb) {
        // ---- stage V block transposed: Vs[hd][key] ----
        {
            int key = kb * KBLK + (tid >> 2);
            int hb  = (tid & 3) * 32;
            const __bf16* vrow = Vb + ((size_t)h * S_LEN + key) * HD + hb;
#pragma unroll
            for (int i = 0; i < 32; i += 8) {
                v8bf t8 = *(const v8bf*)(vrow + i);
#pragma unroll
                for (int j = 0; j < 8; ++j)
                    Vs[hb + i + j][tid >> 2] = t8[j];
            }
        }
        __syncthreads();

        // ---- S = Q * K^T  (K fragments straight from L2-resident global) ----
        v8f sacc[4];
#pragma unroll
        for (int nt = 0; nt < 4; ++nt)
#pragma unroll
            for (int v = 0; v < 8; ++v) sacc[nt][v] = 0.0f;
#pragma unroll
        for (int nt = 0; nt < 4; ++nt) {
            const __bf16* krow =
                Kb + ((size_t)h * S_LEN + kb * KBLK + nt * 16 + lr) * HD;
#pragma unroll
            for (int kt = 0; kt < 4; ++kt) {
                v16bf bfr = load_frag(krow, kt * 32 + half * 8);
                sacc[nt] = __builtin_amdgcn_wmma_f32_16x16x32_bf16(
                    false, qf[kt], false, bfr, (short)0, sacc[nt],
                    false, false);
            }
        }

        // ---- online softmax (rows live in 16-lane halves) ----
        float newm[8], pscale[8], rsum[8];
#pragma unroll
        for (int v = 0; v < 8; ++v) {
            float t = fmaxf(fmaxf(sacc[0][v], sacc[1][v]),
                            fmaxf(sacc[2][v], sacc[3][v]));
#pragma unroll
            for (int m = 1; m < 16; m <<= 1)
                t = fmaxf(t, __shfl_xor(t, m, 32));
            newm[v]   = fmaxf(m_i[v], t);
            pscale[v] = __expf(m_i[v] - newm[v]);
            m_i[v]    = newm[v];
            rsum[v]   = 0.0f;
        }
#pragma unroll
        for (int nt = 0; nt < 4; ++nt)
#pragma unroll
            for (int v = 0; v < 8; ++v) {
                float p = __expf(sacc[nt][v] - newm[v]);
                sacc[nt][v] = p;
                rsum[v] += p;
            }
#pragma unroll
        for (int v = 0; v < 8; ++v) {
#pragma unroll
            for (int m = 1; m < 16; m <<= 1)
                rsum[v] += __shfl_xor(rsum[v], m, 32);
            l_i[v] = l_i[v] * pscale[v] + rsum[v];
        }
#pragma unroll
        for (int t = 0; t < 8; ++t)
#pragma unroll
            for (int v = 0; v < 8; ++v) o[t][v] *= pscale[v];

        // ---- P: C-layout -> A-fragment layout via per-wave LDS tile ----
#pragma unroll
        for (int nt = 0; nt < 4; ++nt)
#pragma unroll
            for (int v = 0; v < 8; ++v)
                Ps[wave][v + 8 * half][nt * 16 + lr] = (__bf16)sacc[nt][v];

        // ---- O += P * V ----
        const __bf16* prow = &Ps[wave][lr][0];
#pragma unroll
        for (int kt2 = 0; kt2 < 2; ++kt2) {
            v16bf pf = load_frag(prow, kt2 * 32 + half * 8);
#pragma unroll
            for (int nt2 = 0; nt2 < 8; ++nt2) {
                v16bf vf = load_frag(&Vs[nt2 * 16 + lr][0],
                                     kt2 * 32 + half * 8);
                o[nt2] = __builtin_amdgcn_wmma_f32_16x16x32_bf16(
                    false, pf, false, vf, (short)0, o[nt2], false, false);
            }
        }
        __syncthreads();   // protect Vs before next block restages it
    }

    // ---- normalize + write bf16 [S][D] with head offset ----
    float inv[8];
#pragma unroll
    for (int v = 0; v < 8; ++v) inv[v] = 1.0f / l_i[v];
#pragma unroll
    for (int nt2 = 0; nt2 < 8; ++nt2)
#pragma unroll
        for (int v = 0; v < 8; ++v) {
            int row = r0 + v + 8 * half;
            int col = h * HD + nt2 * 16 + lr;
            attnb[(size_t)row * DMODEL + col] = (__bf16)(o[nt2][v] * inv[v]);
        }
}

// ---------------------------------------------------------------------------
// Workspace layout (bytes), total 160 MB:
//   qkvb  bf16 4096*6144 :  48 MB @ 0
//   xb    bf16 4096*2048 :  16 MB @  50331648
//   wqb   bf16 2048*6144 :  24 MB @  67108864
//   wob   bf16 2048*2048 :   8 MB @  92274688
//   Qb    bf16 16*4096*128: 16 MB @ 100663296
//   Kb                    : 16 MB @ 117440512
//   Vb                    : 16 MB @ 134217728
//   attnb bf16 4096*2048 : 16 MB @ 150994944
// ---------------------------------------------------------------------------
extern "C" void kernel_launch(void* const* d_in, const int* in_sizes, int n_in,
                              void* d_out, int out_size, void* d_ws,
                              size_t ws_size, hipStream_t stream) {
    const float* x     = (const float*)d_in[0];
    const float* w_qkv = (const float*)d_in[1];
    const float* b_qkv = (const float*)d_in[2];
    const float* w_out = (const float*)d_in[3];
    const float* b_out = (const float*)d_in[4];
    float* out = (float*)d_out;

    char* ws = (char*)d_ws;
    __bf16* qkvb = (__bf16*)(ws);
    __bf16* xb   = (__bf16*)(ws + 50331648ull);
    __bf16* wqb  = (__bf16*)(ws + 67108864ull);
    __bf16* wob  = (__bf16*)(ws + 92274688ull);
    __bf16* Qb   = (__bf16*)(ws + 100663296ull);
    __bf16* Kb   = (__bf16*)(ws + 117440512ull);
    __bf16* Vb   = (__bf16*)(ws + 134217728ull);
    __bf16* attnb= (__bf16*)(ws + 150994944ull);

    // 0) one-time fp32 -> bf16 repacks
    {
        int nx = 4096 * 2048, nwq = 2048 * 6144, nwo = 2048 * 2048;
        to_bf16<<<(nx  + 255) / 256, 256, 0, stream>>>(x,     xb,  nx);
        to_bf16<<<(nwq + 255) / 256, 256, 0, stream>>>(w_qkv, wqb, nwq);
        to_bf16<<<(nwo + 255) / 256, 256, 0, stream>>>(w_out, wob, nwo);
    }

    // 1) qkv = x @ w_qkv + b_qkv      (4096 x 6144 x 2048) -> bf16
    gemm_bias_bf16<__bf16><<<dim3(6144 / 64, 4096 / 128), 256, 0, stream>>>(
        xb, wqb, b_qkv, qkvb, 4096, 6144, 2048);

    // 2) RoPE + head-major repack (+ fold score scale into Q)
    rope_repack<<<(S_LEN * NH * HD) / 256, 256, 0, stream>>>(qkvb, Qb, Kb, Vb);

    // 3) flash attention: 16 heads x 32 query blocks
    attn_kernel<<<NH * (S_LEN / QROWS), 256, 0, stream>>>(Qb, Kb, Vb, attnb);

    // 4) out = attn @ w_out + b_out   (4096 x 2048 x 2048) -> fp32
    gemm_bias_bf16<float><<<dim3(2048 / 64, 4096 / 128), 256, 0, stream>>>(
        attnb, wob, b_out, out, 4096, 2048, 2048);
}
